// DSSA_13116830122705
// MI455X (gfx1250) — matrix operand order repaired and checked by reading, not compiled, not run on hardware
//
#include <hip/hip_runtime.h>

// ---- problem constants -------------------------------------------------
#define TT 4
#define BB 4
#define CC 384
#define HHW 1024
#define NH 12
#define DD 32
#define NP 256
#define KCONV 1536     // C * PATCH * PATCH
#define OC2 768        // 2*C
#define EPSV 1e-5f

typedef __attribute__((ext_vector_type(16))) _Float16 v16h;
typedef __attribute__((ext_vector_type(8)))  _Float16 v8h;
typedef __attribute__((ext_vector_type(8)))  float    v8f;

// ---- WMMA fragment loaders (wave32, 16x16x32 f16) ----------------------
// A operand: 16x32 tile from row-major matrix (lda in elements).
// Lanes 0-15: row M=lane, K=0..7 & 16..23; lanes 16-31: row M=lane-16,
// K=8..15 & 24..31  -> two contiguous 16B loads per lane.
__device__ __forceinline__ v16h frag_a(const _Float16* A, int lda, int lane) {
  int r  = lane & 15;
  int kb = (lane >> 4) * 8;
  const _Float16* p = A + (size_t)r * lda + kb;
  v8h lo = *(const v8h*)(p);
  v8h hi = *(const v8h*)(p + 16);
  v16h o;
#pragma unroll
  for (int i = 0; i < 8; ++i) { o[i] = lo[i]; o[i + 8] = hi[i]; }
  return o;
}

// B operand: 32x16 tile, sourced from B-transposed (row-major N x K, ldb).
// Lane L: N = L%16, K = (L/16)*8 .. +7 & 16+(L/16)*8 .. +7 -> two 16B loads per lane.
__device__ __forceinline__ v16h frag_bt(const _Float16* BT, int ldb, int lane) {
  int n  = lane & 15;
  int kb = (lane >> 4) * 8;
  const _Float16* p = BT + (size_t)n * ldb + kb;
  v8h lo = *(const v8h*)(p);
  v8h hi = *(const v8h*)(p + 16);
  v16h o;
#pragma unroll
  for (int i = 0; i < 8; ++i) { o[i] = lo[i]; o[i + 8] = hi[i]; }
  return o;
}

// ---- CDNA5 async global->LDS copy (16B per lane) -----------------------
__device__ __forceinline__ void async_b128(unsigned lds_off, const _Float16* g) {
  asm volatile("global_load_async_to_lds_b128 %0, %1, off"
               :: "v"(lds_off), "v"(g) : "memory");
}
__device__ __forceinline__ void wait_async_le2() {
  asm volatile("s_wait_asynccnt 0x2" ::: "memory");
}
__device__ __forceinline__ void wait_async_0() {
  asm volatile("s_wait_asynccnt 0x0" ::: "memory");
}

// ---- K1: LIF over x; emit spikes in xsT (hw-major) and im2colT ---------
extern "C" __global__ void k1_lif_x(const float* __restrict__ x,
                                    _Float16* __restrict__ xsT,
                                    _Float16* __restrict__ im2,
                                    int n1) {
  int i = blockIdx.x * blockDim.x + threadIdx.x;
  if (i >= n1) return;
  int b  = i / (CC * HHW);
  int r  = i - b * (CC * HHW);
  int c  = r / HHW;
  int hw = r - c * HHW;
  int h = hw >> 5, w = hw & 31;
  int p = (h >> 1) * 16 + (w >> 1);        // patch index
  int q = (h & 1) * 2 + (w & 1);           // within-patch index
  int k = c * 4 + q;                       // im2col K index
  float v = 0.0f;
#pragma unroll
  for (int t = 0; t < TT; ++t) {
    float xv = x[(size_t)t * n1 + i];
    v = 0.5f * (v + xv);                   // v += (x - v)/TAU, TAU=2
    float s = (v >= 1.0f) ? 1.0f : 0.0f;   // heaviside(v - 1)
    v *= (1.0f - s);                       // reset
    _Float16 sh = (_Float16)s;
    xsT[(((size_t)t * BB + b) * HHW + hw) * CC + c] = sh;
    im2[(((size_t)t * BB + b) * NP + p) * KCONV + k] = sh;
  }
}

// ---- K2: fp32 -> f16 weight conversion ---------------------------------
extern "C" __global__ void k2_cvt(const float* __restrict__ Wconv,
                                  const float* __restrict__ Wproj,
                                  _Float16* __restrict__ WcH,
                                  _Float16* __restrict__ WpH) {
  int i = blockIdx.x * blockDim.x + threadIdx.x;
  const int NC = OC2 * KCONV;              // 1179648
  const int NPJ = CC * CC;                 // 147456
  if (i < NC)            WcH[i] = (_Float16)Wconv[i];
  else if (i < NC + NPJ) WpH[i - NC] = (_Float16)Wproj[i - NC];
}

// ---- K3: conv-as-GEMM (M=768,N=256,K=1536) + BN1 -> y1T, y2 ------------
// 256 threads (8 waves) per block; 64x64 output tile; K staged through LDS
// in 32-wide steps via async global->LDS copies, double buffered.
extern "C" __global__ void k3_conv(const _Float16* __restrict__ Wc,
                                   const _Float16* __restrict__ im2,
                                   const float* __restrict__ gamma1,
                                   const float* __restrict__ beta1,
                                   _Float16* __restrict__ y1T,
                                   _Float16* __restrict__ y2) {
  __shared__ __align__(16) _Float16 Abuf[2][64 * 32];  // 4KB each
  __shared__ __align__(16) _Float16 Bbuf[2][64 * 32];  // 4KB each
  int tid  = threadIdx.x;
  int lane = tid & 31;
  int wave = tid >> 5;
  int bid  = blockIdx.x;
  int tb   = bid / (12 * 4);
  int rem  = bid - tb * (12 * 4);
  int mt   = rem >> 2;                     // 0..11 (64 rows of OC2)
  int nt   = rem & 3;                      // 0..3  (64 cols of NP)
  const _Float16* Ag  = Wc  + (size_t)(mt * 64) * KCONV;
  const _Float16* BTg = im2 + (size_t)tb * NP * KCONV + (size_t)(nt * 64) * KCONV;
  int row = tid >> 2;                      // 0..63
  int chk = tid & 3;                       // 0..3 (16B chunk in 64B row)

  // stage s: copy A/B 64x32 panels (k0 = s*32) into Abuf/Bbuf[s&1]
  auto issue = [&](int s) {
    int k0 = s * 32;
    const _Float16* ga = Ag  + (size_t)row * KCONV + k0 + chk * 8;
    const _Float16* gb = BTg + (size_t)row * KCONV + k0 + chk * 8;
    unsigned la = (unsigned)(size_t)&Abuf[s & 1][row * 32 + chk * 8];
    unsigned lb = (unsigned)(size_t)&Bbuf[s & 1][row * 32 + chk * 8];
    async_b128(la, ga);
    async_b128(lb, gb);
  };

  int mrow = wave & 3;                     // M subtile 0..3
  int nb   = (wave >> 2) * 2;              // N subtile base: 0 or 2
  v8f acc0 = {}, acc1 = {};

  issue(0);
  for (int s = 0; s < 48; ++s) {
    if (s + 1 < 48) { issue(s + 1); wait_async_le2(); }
    else            { wait_async_0(); }
    __syncthreads();
    const _Float16* As = &Abuf[s & 1][0];
    const _Float16* Bs = &Bbuf[s & 1][0];
    v16h a  = frag_a (As + mrow * 16 * 32, 32, lane);
    v16h b0 = frag_bt(Bs + (nb + 0) * 16 * 32, 32, lane);
    v16h b1 = frag_bt(Bs + (nb + 1) * 16 * 32, 32, lane);
    acc0 = __builtin_amdgcn_wmma_f32_16x16x32_f16(false, a, false, b0,
                                                  (short)0, acc0, false, false);
    acc1 = __builtin_amdgcn_wmma_f32_16x16x32_f16(false, a, false, b1,
                                                  (short)0, acc1, false, false);
    __syncthreads();
  }

  int nl  = lane & 15;
  int mh8 = (lane >> 4) * 8;
  float rs = rsqrtf(1.0f + EPSV);
#pragma unroll
  for (int sub = 0; sub < 2; ++sub) {
    v8f acc = sub ? acc1 : acc0;
#pragma unroll
    for (int i = 0; i < 8; ++i) {
      int o = mt * 64 + mrow * 16 + i + mh8;       // output channel (M)
      int p = nt * 64 + (nb + sub) * 16 + nl;      // patch (N)
      float val = acc[i] * (gamma1[o] * rs) + beta1[o];
      int h = o >> 6, oo = o & 63;
      if (oo < DD)   // y1 stored transposed (Np x d): A-operand of attn GEMM
        y1T[(((size_t)tb * NH + h) * NP + p) * DD + oo] = (_Float16)val;
      else           // y2 row-major (d x Np): B^T-operand of out GEMM
        y2 [(((size_t)tb * NH + h) * DD + (oo - DD)) * NP + p] = (_Float16)val;
    }
  }
}

// ---- K4: fused attn GEMM -> LIF -> out GEMM -> LIF over t --------------
// block = 512 thr (16 waves); grid = B*NH*64 (one 16-wide m-tile per block)
extern "C" __global__ void k4_attn(const _Float16* __restrict__ y1T,
                                   const _Float16* __restrict__ y2,
                                   const _Float16* __restrict__ xsT,
                                   const float* __restrict__ frx,
                                   const float* __restrict__ fra,
                                   _Float16* __restrict__ outS) {
  __shared__ __align__(16) _Float16 attn_sT[16 * NP];   // [m_local][n], 8KB
  int lane = threadIdx.x & 31;
  int wave = threadIdx.x >> 5;             // n-tile owner (0..15)
  int bh = blockIdx.x >> 6;                // / 64
  int mt = blockIdx.x & 63;
  int b = bh / NH, h = bh - b * NH;
  int m0 = mt * 16;
  float scale1 = rsqrtf(frx[h] * (float)DD);
  float scale2 = rsqrtf(fra[h] * (float)NP);
  int nl  = lane & 15;
  int mh8 = (lane >> 4) * 8;
  v8f va = {};                             // attn membrane (per n-tile wave)
  v8f vo = {};                             // out membrane (waves 0,1)
  for (int t = 0; t < TT; ++t) {
    size_t tbh = ((size_t)t * BB + b) * NH + h;
    // GEMM1: attn[n,m] = sum_d y1T[n,d] * xr[d,m]  (K = 32, one WMMA)
    const _Float16* A  = y1T + tbh * (NP * DD) + (size_t)wave * 16 * DD;
    const _Float16* BT = xsT + ((size_t)t * BB + b) * (HHW * CC)
                             + (size_t)m0 * CC + h * DD;
    v16h a  = frag_a (A, DD, lane);
    v16h bf = frag_bt(BT, CC, lane);
    v8f c = {};
    c = __builtin_amdgcn_wmma_f32_16x16x32_f16(false, a, false, bf,
                                               (short)0, c, false, false);
#pragma unroll
    for (int i = 0; i < 8; ++i) {          // LIF on attn, spikes -> LDS^T
      float pre = c[i] * scale1;
      float vv  = 0.5f * (va[i] + pre);
      float s   = (vv >= 1.0f) ? 1.0f : 0.0f;
      va[i] = vv * (1.0f - s);
      int n_g = wave * 16 + i + mh8;       // M index = n
      attn_sT[nl * NP + n_g] = (_Float16)s;
    }
    __syncthreads();
    if (wave < 2) {
      // GEMM2: outT[m,d] = sum_n attnT[m,n] * y2[d,n]  (K = 256)
      const _Float16* B2 = y2 + tbh * (DD * NP) + (size_t)wave * 16 * NP;
      v8f c2 = {};
      for (int kk = 0; kk < 8; ++kk) {
        v16h a2 = frag_a (attn_sT + kk * 32, NP, lane);
        v16h b2 = frag_bt(B2      + kk * 32, NP, lane);
        c2 = __builtin_amdgcn_wmma_f32_16x16x32_f16(false, a2, false, b2,
                                                    (short)0, c2, false, false);
      }
#pragma unroll
      for (int i = 0; i < 8; ++i) {        // LIF on out, spikes -> global
        float pre = c2[i] * scale2;
        float vv  = 0.5f * (vo[i] + pre);
        float s   = (vv >= 1.0f) ? 1.0f : 0.0f;
        vo[i] = vv * (1.0f - s);
        int m_g = m0 + i + mh8;            // M index = m
        int d_g = wave * 16 + nl;          // N index = d
        outS[(((size_t)t * BB + b) * HHW + m_g) * CC + h * DD + d_g]
            = (_Float16)s;
      }
    }
    __syncthreads();                       // protect LDS before next t
  }
}

// ---- K5: projection GEMM (M=1024,N=384,K=384) + BN2 + residual ---------
extern "C" __global__ void k5_proj(const _Float16* __restrict__ outS,
                                   const _Float16* __restrict__ Wp,
                                   const float* __restrict__ gamma2,
                                   const float* __restrict__ beta2,
                                   const float* __restrict__ x,
                                   float* __restrict__ out) {
  int lane = threadIdx.x & 31;
  int bid  = blockIdx.x;
  int tb   = bid / (64 * 24);
  int rem  = bid - tb * (64 * 24);
  int mt   = rem / 24;
  int nt   = rem - mt * 24;
  const _Float16* A  = outS + (size_t)tb * (HHW * CC) + (size_t)mt * 16 * CC;
  const _Float16* BT = Wp + (size_t)nt * 16 * CC;      // Wproj row-major o x c
  v8f acc = {};
  for (int kk = 0; kk < 12; ++kk) {
    if (kk + 1 < 12) {
      __builtin_prefetch(A  + (size_t)(kk + 1) * 32 + (size_t)(lane & 15) * CC, 0, 3);
      __builtin_prefetch(BT + (size_t)(kk + 1) * 32 + (size_t)(lane & 15) * CC, 0, 3);
    }
    v16h a  = frag_a (A  + kk * 32, CC, lane);
    v16h bf = frag_bt(BT + kk * 32, CC, lane);
    acc = __builtin_amdgcn_wmma_f32_16x16x32_f16(false, a, false, bf,
                                                 (short)0, acc, false, false);
  }
  int nl  = lane & 15;
  int mh8 = (lane >> 4) * 8;
  float rs = rsqrtf(1.0f + EPSV);
#pragma unroll
  for (int i = 0; i < 8; ++i) {
    int m = mt * 16 + i + mh8;             // hw position (M)
    int o = nt * 16 + nl;                  // channel (N)
    size_t idx = ((size_t)tb * CC + o) * HHW + m;
    out[idx] = acc[i] * (gamma2[o] * rs) + beta2[o] + x[idx];
  }
}

// ---- workspace layout (bytes) ------------------------------------------
#define WS_XST   ((size_t)0)                               // 12,582,912
#define WS_IM2   (WS_XST + (size_t)TT*BB*HHW*CC*2)         // 12,582,912
#define WS_WC    (WS_IM2 + (size_t)TT*BB*NP*KCONV*2)       //  2,359,296
#define WS_WP    (WS_WC  + (size_t)OC2*KCONV*2)            //    294,912
#define WS_Y1T   (WS_WP  + (size_t)CC*CC*2)                //  3,145,728
#define WS_Y2    (WS_Y1T + (size_t)TT*BB*NH*NP*DD*2)       //  3,145,728
#define WS_OUTS  (WS_Y2  + (size_t)TT*BB*NH*DD*NP*2)       // 12,582,912

extern "C" void kernel_launch(void* const* d_in, const int* in_sizes, int n_in,
                              void* d_out, int out_size, void* d_ws, size_t ws_size,
                              hipStream_t stream) {
  const float* x      = (const float*)d_in[0];
  const float* Wconv  = (const float*)d_in[1];
  const float* gamma1 = (const float*)d_in[2];
  const float* beta1  = (const float*)d_in[3];
  const float* Wproj  = (const float*)d_in[4];
  const float* gamma2 = (const float*)d_in[5];
  const float* beta2  = (const float*)d_in[6];
  const float* frx    = (const float*)d_in[7];
  const float* fra    = (const float*)d_in[8];
  float* out = (float*)d_out;
  char* ws = (char*)d_ws;

  _Float16* xsT  = (_Float16*)(ws + WS_XST);
  _Float16* im2  = (_Float16*)(ws + WS_IM2);
  _Float16* WcH  = (_Float16*)(ws + WS_WC);
  _Float16* WpH  = (_Float16*)(ws + WS_WP);
  _Float16* y1T  = (_Float16*)(ws + WS_Y1T);
  _Float16* y2   = (_Float16*)(ws + WS_Y2);
  _Float16* outS = (_Float16*)(ws + WS_OUTS);

  const int n1 = BB * CC * HHW;                       // 1,572,864
  k1_lif_x<<<(n1 + 255) / 256, 256, 0, stream>>>(x, xsT, im2, n1);

  const int n2 = OC2 * KCONV + CC * CC;               // 1,327,104
  k2_cvt<<<(n2 + 255) / 256, 256, 0, stream>>>(Wconv, Wproj, WcH, WpH);

  k3_conv<<<TT * BB * 12 * 4, 256, 0, stream>>>(WcH, im2, gamma1, beta1,
                                                y1T, y2);

  k4_attn<<<BB * NH * 64, 512, 0, stream>>>(y1T, y2, xsT, frx, fra, outS);

  k5_proj<<<TT * BB * 64 * 24, 32, 0, stream>>>(outS, WpH, gamma2, beta2,
                                                x, out);
}
